// Recurrent_37726992728146
// MI455X (gfx1250) — matrix operand order repaired and checked by reading, not compiled, not run on hardware
//
#include <hip/hip_runtime.h>
#include <hip/hip_bf16.h>

typedef __attribute__((ext_vector_type(16))) _Float16 v16h;
typedef __attribute__((ext_vector_type(8)))  _Float16 v8h;
typedef __attribute__((ext_vector_type(8)))  float    v8f;

#define B_  32
#define T_  512
#define D_  512
#define H_  1024
#define SCAN_BLOCKS 32u

// ---------------------------------------------------------------------------
// Pack a row-major f32 [K x N] weight matrix into f16 WMMA-B fragments.
// Per 32x16 (KxN) tile (ISA 7.12.2, 16-bit B layout): lane l holds 16
// halfs for column n = nt*16 + (l&15), rows k = kt*32 + (l>>4)*8 + i (+8 for i>=8).
// ---------------------------------------------------------------------------
__global__ void pack_b_kernel(const float* __restrict__ W, _Float16* __restrict__ P,
                              int K, int N) {
  int tid    = blockIdx.x * blockDim.x + threadIdx.x;   // one thread per (tile, lane)
  int ntiles = N >> 4;
  int total  = (K >> 5) * ntiles * 32;
  if (tid >= total) return;
  int lane = tid & 31;
  int tile = tid >> 5;
  int nt   = tile % ntiles;
  int kt   = tile / ntiles;
  int n  = (nt << 4) + (lane & 15);
  int k0 = (kt << 5) + ((lane >> 4) << 3);
  v16h v;
#pragma unroll
  for (int i = 0; i < 16; ++i) v[i] = (_Float16)W[(size_t)(k0 + i + ((i >> 3) << 3)) * N + n];
  *(v16h*)(P + (size_t)tid * 16) = v;
}

__global__ void cvt_h0_kernel(const float* __restrict__ h0, _Float16* __restrict__ h16) {
  int tid = blockIdx.x * blockDim.x + threadIdx.x;
  if (tid < B_ * H_) h16[tid] = (_Float16)h0[tid];
}

__global__ void init_sync_kernel(unsigned* p) {
  if (threadIdx.x == 0) { p[0] = 0u; p[1] = 0u; }
}

// ---------------------------------------------------------------------------
// Phase 1: xw = x @ Wx + b  ->  written as f32 pre-activations into d_out.
// Each wave: one 16(M) x 64(N) strip, K=512 in 16 steps of wmma 16x16x32 f16.
// Software-pipelined one step ahead and FULLY unrolled so the register
// allocator renames prefetch buffers per step (no rotation copies).
// ---------------------------------------------------------------------------
__global__ __launch_bounds__(128)
void gemm_xw_kernel(const float* __restrict__ x, const _Float16* __restrict__ WxP,
                    const float* __restrict__ bias, float* __restrict__ out) {
  const int lane = threadIdx.x & 31;
  const int wave = threadIdx.x >> 5;
  const int mt   = blockIdx.x * 4 + wave;   // [0, 1024)  (M = 16384 / 16)
  const int nt0  = blockIdx.y * 4;          // [0, 64) step 4
  const int m    = (mt << 4) + (lane & 15);
  const int kb   = (lane >> 4) << 3;        // A-frag K sub-offset: 0 or 8

  const float*    arow = x + (size_t)m * D_ + kb;
  const _Float16* bcol = WxP + ((size_t)nt0 * 32 + lane) * 16;

  v8f acc[4] = {};
  // Prologue: fetch k-step 0 operands.
  v8f f0 = *(const v8f*)(arow);
  v8f f1 = *(const v8f*)(arow + 16);
  v16h b0 = *(const v16h*)(bcol);
  v16h b1 = *(const v16h*)(bcol + 512);
  v16h b2 = *(const v16h*)(bcol + 1024);
  v16h b3 = *(const v16h*)(bcol + 1536);

#pragma unroll
  for (int kt = 0; kt < 16; ++kt) {
    v16h a;
#pragma unroll
    for (int i = 0; i < 8; ++i) { a[i] = (_Float16)f0[i]; a[i + 8] = (_Float16)f1[i]; }
    v16h c0 = b0, c1 = b1, c2 = b2, c3 = b3;
    if (kt < 15) {                 // prefetch k-step kt+1 while WMMAs execute
      const float*    ap = arow + ((kt + 1) << 5);
      const _Float16* bp = bcol + (size_t)(kt + 1) * 64 * 512;
      f0 = *(const v8f*)(ap);
      f1 = *(const v8f*)(ap + 16);
      b0 = *(const v16h*)(bp);
      b1 = *(const v16h*)(bp + 512);
      b2 = *(const v16h*)(bp + 1024);
      b3 = *(const v16h*)(bp + 1536);
    }
    acc[0] = __builtin_amdgcn_wmma_f32_16x16x32_f16(false, a, false, c0, (short)0, acc[0], false, false);
    acc[1] = __builtin_amdgcn_wmma_f32_16x16x32_f16(false, a, false, c1, (short)0, acc[1], false, false);
    acc[2] = __builtin_amdgcn_wmma_f32_16x16x32_f16(false, a, false, c2, (short)0, acc[2], false, false);
    acc[3] = __builtin_amdgcn_wmma_f32_16x16x32_f16(false, a, false, c3, (short)0, acc[3], false, false);
  }

  const int ncol = lane & 15;
  const int mrow = (mt << 4) + ((lane >> 4) << 3);   // C/D layout: M = r + 8*(lane>>4)
#pragma unroll
  for (int j = 0; j < 4; ++j) {
    int n = ((nt0 + j) << 4) + ncol;
    float bb = bias[n];
#pragma unroll
    for (int r = 0; r < 8; ++r)
      out[(size_t)(mrow + r) * H_ + n] = acc[j][r] + bb;
  }
}

// ---------------------------------------------------------------------------
// Grid-wide barrier (two-variable sense barrier). All SCAN_BLOCKS blocks are
// co-resident (32 blocks x 160KB LDS -> 2 blocks/WGP, needs only 16 WGPs).
// ---------------------------------------------------------------------------
__device__ inline void grid_sync(unsigned* cnt, unsigned* gen, unsigned nblocks) {
  __syncthreads();
  if (threadIdx.x == 0) {
    __threadfence();   // make this step's h stores device-visible
    unsigned g = __hip_atomic_load(gen, __ATOMIC_RELAXED, __HIP_MEMORY_SCOPE_AGENT);
    unsigned prev = __hip_atomic_fetch_add(cnt, 1u, __ATOMIC_ACQ_REL, __HIP_MEMORY_SCOPE_AGENT);
    if (prev == nblocks - 1u) {
      __hip_atomic_store(cnt, 0u, __ATOMIC_RELAXED, __HIP_MEMORY_SCOPE_AGENT);
      __hip_atomic_store(gen, g + 1u, __ATOMIC_RELEASE, __HIP_MEMORY_SCOPE_AGENT);
    } else {
      while (__hip_atomic_load(gen, __ATOMIC_ACQUIRE, __HIP_MEMORY_SCOPE_AGENT) == g)
        __builtin_amdgcn_s_sleep(1);
    }
  }
  __syncthreads();
}

// ---------------------------------------------------------------------------
// Phase 2: persistent recurrence. 32 blocks of 4 waves; block owns a fixed
// (mtile, 4-ntile) patch of h. Its 128KB slice of packed Wh lives in LDS for
// all 512 steps; the block's 32KB h-slice is staged into LDS each step and
// shared by all 4 waves. Four rotating accumulators break the serial
// WMMA dependence chain. h double-buffered in f16 in global memory.
// io holds xw pre-activations and is overwritten in place with tanh outputs.
// ---------------------------------------------------------------------------
__global__ __launch_bounds__(128)
void rnn_scan_kernel(float* __restrict__ io, _Float16* __restrict__ hA,
                     _Float16* __restrict__ hB, const _Float16* __restrict__ WhP,
                     float* __restrict__ last, unsigned* cnt, unsigned* gen) {
  extern __shared__ _Float16 smem[];
  _Float16* whs = smem;            // 65536 halfs = 128 KB: packed Wh slice
  _Float16* hs  = smem + 65536;    // 16384 halfs =  32 KB: this mtile's h rows
  const int tid  = threadIdx.x;
  const int lane = tid & 31;
  const int wave = tid >> 5;
  const int mt   = (int)blockIdx.x & 1;                 // 2 M-tiles (B = 32)
  const int nt   = (((int)blockIdx.x) >> 1) * 4 + wave; // [0, 64)

  // Preload this block's Wh fragments into LDS (once, wave-private tiles).
  for (int kt = 0; kt < 32; ++kt) {
    const _Float16* src = WhP + ((size_t)(kt * 64 + nt) * 32 + lane) * 16;
    _Float16*       dst = whs + ((size_t)((kt << 2) + wave) * 32 + lane) * 16;
    *(v16h*)dst = *(const v16h*)src;
  }

  const int mloc = lane & 15;                          // A-frag local row
  const int kb   = (lane >> 4) << 3;                   // A-frag K sub-offset
  const int n    = (nt << 4) + (lane & 15);            // C/D column
  const int mrow = (mt << 4) + ((lane >> 4) << 3);     // C/D row base (batch)

  for (int t = 0; t < T_; ++t) {
    const _Float16* hc = (t & 1) ? hB : hA;
    _Float16*       hn = (t & 1) ? hA : hB;

    // Stage h rows [mt*16, mt*16+16) x H into LDS (shared by the 4 waves).
    {
      const _Float16* src = hc + (size_t)mt * 16 * H_;
#pragma unroll
      for (int i = 0; i < 8; ++i) {
        int c = i * 128 + tid;                         // 1024 chunks of 16 halfs
        *(v16h*)(hs + (size_t)c * 16) = *(const v16h*)(src + (size_t)c * 16);
      }
    }
    // Prefetch xw pre-activations now so the loads overlap the WMMA loop.
    float pre[8];
#pragma unroll
    for (int r = 0; r < 8; ++r)
      pre[r] = io[((size_t)(mrow + r) * T_ + t) * H_ + n];
    __syncthreads();

    // 4 rotating accumulators -> 4 independent WMMA chains of depth 8.
    v8f accv[4] = {};
#pragma unroll
    for (int kt = 0; kt < 32; ++kt) {
      const _Float16* ap = hs + (size_t)mloc * H_ + (kt << 5) + kb;
      v8h l0 = *(const v8h*)ap;
      v8h l1 = *(const v8h*)(ap + 16);
      v16h a;
#pragma unroll
      for (int i = 0; i < 8; ++i) { a[i] = l0[i]; a[i + 8] = l1[i]; }
      const _Float16* bp = whs + ((size_t)((kt << 2) + wave) * 32 + lane) * 16;
      v16h b = *(const v16h*)bp;
      accv[kt & 3] = __builtin_amdgcn_wmma_f32_16x16x32_f16(
          false, a, false, b, (short)0, accv[kt & 3], false, false);
    }
    v8f acc = (accv[0] + accv[1]) + (accv[2] + accv[3]);

#pragma unroll
    for (int r = 0; r < 8; ++r) {
      int row = mrow + r;                              // batch index
      size_t idx = ((size_t)row * T_ + t) * H_ + n;
      float h = tanhf(pre[r] + acc[r]);
      io[idx] = h;                                     // output[b, t, h]
      hn[(size_t)row * H_ + n] = (_Float16)h;          // next-step state (f16)
      if (t == T_ - 1) last[(size_t)row * H_ + n] = h; // last_state
    }
    grid_sync(cnt, gen, SCAN_BLOCKS);
  }
}

// ---------------------------------------------------------------------------
extern "C" void kernel_launch(void* const* d_in, const int* in_sizes, int n_in,
                              void* d_out, int out_size, void* d_ws, size_t ws_size,
                              hipStream_t stream) {
  (void)in_sizes; (void)n_in; (void)out_size; (void)ws_size;
  const float* x  = (const float*)d_in[0];   // [B,T,D]
  const float* h0 = (const float*)d_in[1];   // [B,H]
  const float* Wx = (const float*)d_in[2];   // [D,H]
  const float* Wh = (const float*)d_in[3];   // [H,H]
  const float* bv = (const float*)d_in[4];   // [H]
  float* out = (float*)d_out;                // [B,T,H] outputs ++ [B,H] last_state

  char* ws = (char*)d_ws;
  _Float16* WhP = (_Float16*)ws;                                   // 2 MB
  _Float16* WxP = (_Float16*)(ws + (2u << 20));                    // 1 MB
  _Float16* hA  = (_Float16*)(ws + (3u << 20));                    // 64 KB
  _Float16* hB  = (_Float16*)(ws + (3u << 20) + (1u << 16));       // 64 KB
  unsigned* sync = (unsigned*)(ws + (3u << 20) + (2u << 16));      // 8 B

  // Pack weights into WMMA-B fragment layout (f16).
  pack_b_kernel<<<256, 256, 0, stream>>>(Wh, WhP, H_, H_);   // 32*64*32 = 65536 thr
  pack_b_kernel<<<128, 256, 0, stream>>>(Wx, WxP, D_, H_);   // 16*64*32 = 32768 thr
  cvt_h0_kernel<<<(B_ * H_ + 255) / 256, 256, 0, stream>>>(h0, hA);
  init_sync_kernel<<<1, 32, 0, stream>>>(sync);

  // Phase 1: xw = x@Wx + b -> d_out (pre-activations).
  gemm_xw_kernel<<<dim3(256, 16), 128, 0, stream>>>(x, WxP, bv, out);

  // Phase 2: persistent scan over T=512 steps (160 KB dynamic LDS per block).
  float* last = out + (size_t)B_ * T_ * H_;
  rnn_scan_kernel<<<SCAN_BLOCKS, 128, 163840, stream>>>(out, hA, hB, WhP, last,
                                                        sync, sync + 1);
}
